// CausalAttention_28338194219367
// MI455X (gfx1250) — compile-verified, matrix operand order checked by reading, not run on hardware
//
#include <hip/hip_runtime.h>
#include <hip/hip_bf16.h>
#include <stdint.h>

// Problem constants (match reference)
constexpr int BB = 4;
constexpr int SS = 2048;
constexpr int DD = 1024;          // D_IN == D_OUT
constexpr float SCALE = 0.03125f; // 1024^-0.5

typedef __attribute__((ext_vector_type(16))) __bf16    v16bf;
typedef __attribute__((ext_vector_type(8)))  float     v8f;
typedef __attribute__((ext_vector_type(8)))  unsigned  v8u;

union Frag { v8u u; v16bf bf; };

__device__ __forceinline__ unsigned short f32_to_bf16(float f) {
  unsigned u = __builtin_bit_cast(unsigned, f);
  u += 0x7FFFu + ((u >> 16) & 1u);           // round-to-nearest-even
  return (unsigned short)(u >> 16);
}

// CDNA5 async global->LDS copy (GLOBAL_LOAD_ASYNC_TO_LDS_B128, GV mode).
// %0 = LDS byte address (VGPR), %1 = 64-bit global address (VGPR pair).
__device__ __forceinline__ void async_b128(unsigned lds_addr, const void* gaddr) {
  asm volatile("global_load_async_to_lds_b128 %0, %1, off"
               :: "v"(lds_addr), "v"(gaddr) : "memory");
}
__device__ __forceinline__ void wait_async0() {
  asm volatile("s_wait_asynccnt 0x0" ::: "memory");
}
__device__ __forceinline__ unsigned lds_off(const void* p) {
  // Generic (flat) LDS pointers carry the LDS byte offset in addr[31:0].
  return (unsigned)(uintptr_t)p;
}

// ---------------------------------------------------------------------------
// 1a) f32 -> bf16 conversion (flat)
// ---------------------------------------------------------------------------
__global__ void cvt_f32_bf16_kernel(const float* __restrict__ src,
                                    unsigned short* __restrict__ dst, int n) {
  int i = blockIdx.x * blockDim.x + threadIdx.x;
  if (i < n) dst[i] = f32_to_bf16(src[i]);
}

// ---------------------------------------------------------------------------
// 1b) f32 -> bf16 transposing conversion: dst[c][r] = bf16(src[r][c])
//     block (32,8), grid (C/32, R/32)
// ---------------------------------------------------------------------------
__global__ void cvt_transpose_bf16_kernel(const float* __restrict__ src,
                                          unsigned short* __restrict__ dst,
                                          int R, int C) {
  __shared__ unsigned short t[32][33];
  const int c0 = blockIdx.x * 32, r0 = blockIdx.y * 32;
  for (int rr = threadIdx.y; rr < 32; rr += 8)
    t[rr][threadIdx.x] = f32_to_bf16(src[(size_t)(r0 + rr) * C + c0 + threadIdx.x]);
  __syncthreads();
  for (int cc = threadIdx.y; cc < 32; cc += 8)
    dst[(size_t)(c0 + cc) * R + r0 + threadIdx.x] = t[threadIdx.x][cc];
}

// ---------------------------------------------------------------------------
// 1c) bf16 -> bf16 batched transpose: dst[b][c][r] = src[b*R + r][c]
//     block (32,8), grid (C/32, R/32, B)
// ---------------------------------------------------------------------------
__global__ void transpose_bf16_kernel(const unsigned short* __restrict__ src,
                                      unsigned short* __restrict__ dst,
                                      int R, int C) {
  __shared__ unsigned short t[32][33];
  const int b = blockIdx.z;
  const int c0 = blockIdx.x * 32, r0 = blockIdx.y * 32;
  const size_t sb = (size_t)b * R * C;
  for (int rr = threadIdx.y; rr < 32; rr += 8)
    t[rr][threadIdx.x] = src[sb + (size_t)(r0 + rr) * C + c0 + threadIdx.x];
  __syncthreads();
  for (int cc = threadIdx.y; cc < 32; cc += 8)
    dst[sb + (size_t)(c0 + cc) * R + r0 + threadIdx.x] = t[threadIdx.x][cc];
}

// ---------------------------------------------------------------------------
// 2) bf16 GEMM: C[M,N] = A[M,K] * WT[N,K]^T, output bf16.  Block tile 128x64,
//    256 threads = 8 waves, each wave: 16 rows x 64 cols (4 wmma accums).
// ---------------------------------------------------------------------------
__global__ __launch_bounds__(256)
void gemm_bf16_kernel(const unsigned short* __restrict__ A,
                      const unsigned short* __restrict__ WT,
                      unsigned short* __restrict__ C,
                      int M, int N, int K) {
  __shared__ __align__(16) unsigned int Atile[128 * 16];  // [128 m][16 u32] 8 KB
  __shared__ __align__(16) unsigned int Btile[64 * 16];   // [64 n][16 u32]  4 KB

  const int tid  = threadIdx.x;
  const int wave = tid >> 5;
  const int lane = tid & 31;
  const int half = lane >> 4;
  const int ln   = lane & 15;
  const int m0   = blockIdx.y * 128;
  const int n0   = blockIdx.x * 64;

  const unsigned int* A32 = (const unsigned int*)A;
  const unsigned int* W32 = (const unsigned int*)WT;

  v8f acc[4] = {};

  for (int kc = 0; kc < K; kc += 32) {
#pragma unroll
    for (int it = 0; it < 2; ++it) {
      int i = (it * 256 + tid) * 4;                 // u32 index
      int row = i >> 4, col = i & 15;
      async_b128(lds_off(&Atile[i]),
                 &A32[(size_t)(m0 + row) * (K >> 1) + (kc >> 1) + col]);
    }
    {
      int i = tid * 4;
      int row = i >> 4, col = i & 15;
      async_b128(lds_off(&Btile[i]),
                 &W32[(size_t)(n0 + row) * (K >> 1) + (kc >> 1) + col]);
    }
    if (kc + 32 < K)
      __builtin_prefetch(&A32[(size_t)(m0 + (tid >> 1)) * (K >> 1) + ((kc + 32) >> 1)], 0, 0);
    wait_async0();
    __syncthreads();

    Frag a;
    const unsigned int* Arow = &Atile[(wave * 16 + ln) * 16];
#pragma unroll
    for (int j = 0; j < 4; ++j) {
      a.u[j]     = Arow[half * 4 + j];
      a.u[4 + j] = Arow[8 + half * 4 + j];
    }
#pragma unroll
    for (int t = 0; t < 4; ++t) {
      Frag b;
      const unsigned int* Brow = &Btile[(t * 16 + ln) * 16];
#pragma unroll
      for (int j = 0; j < 4; ++j) {
        b.u[j]     = Brow[half * 4 + j];
        b.u[4 + j] = Brow[8 + half * 4 + j];
      }
      acc[t] = __builtin_amdgcn_wmma_f32_16x16x32_bf16(
          false, a.bf, false, b.bf, (short)0, acc[t], false, false);
    }
    __syncthreads();
  }

#pragma unroll
  for (int t = 0; t < 4; ++t)
#pragma unroll
    for (int r = 0; r < 8; ++r) {
      int row = m0 + wave * 16 + r + 8 * half;
      int col = n0 + t * 16 + ln;
      C[(size_t)row * N + col] = f32_to_bf16(acc[t][r]);
    }
}

// ---------------------------------------------------------------------------
// 3) Pass 1: per-row softmax stats over causal scores.
// ---------------------------------------------------------------------------
__global__ __launch_bounds__(256)
void attn_stats_kernel(const unsigned short* __restrict__ Q,
                       const unsigned short* __restrict__ Km,
                       float* __restrict__ mbuf, float* __restrict__ lbuf) {
  __shared__ __align__(16) unsigned int Kt[32 * (DD / 2)];  // 64 KB

  const int tid  = threadIdx.x;
  const int wave = tid >> 5;
  const int lane = tid & 31;
  const int half = lane >> 4;
  const int ln   = lane & 15;
  const int b    = blockIdx.y;
  const int q0   = blockIdx.x * 128;

  const int qrow = q0 + wave * 16 + ln;
  const unsigned int* Qrow = (const unsigned int*)Q + ((size_t)b * SS + qrow) * (DD / 2);

  float mreg[8], lreg[8];
#pragma unroll
  for (int r = 0; r < 8; ++r) { mreg[r] = -__builtin_inff(); lreg[r] = 0.f; }

  const int nkv = ((q0 + 127) >> 5) + 1;
  for (int kt = 0; kt < nkv; ++kt) {
    const int kv0 = kt * 32;
    const unsigned int* Kg = (const unsigned int*)Km + ((size_t)b * SS + kv0) * (DD / 2);
#pragma unroll
    for (int it = 0; it < 16; ++it) {
      int i = (it * 256 + tid) * 4;
      async_b128(lds_off(&Kt[i]), Kg + i);
    }
    wait_async0();
    __syncthreads();

    v8f sacc[2] = {};
    for (int kc = 0; kc < DD; kc += 32) {
      Frag a;
      const int c0 = (kc >> 1) + half * 4;
      const int c1 = (kc >> 1) + 8 + half * 4;
#pragma unroll
      for (int j = 0; j < 4; ++j) { a.u[j] = Qrow[c0 + j]; a.u[4 + j] = Qrow[c1 + j]; }
#pragma unroll
      for (int t = 0; t < 2; ++t) {
        Frag bfr;
        const unsigned int* Krow = &Kt[(t * 16 + ln) * (DD / 2)];
#pragma unroll
        for (int j = 0; j < 4; ++j) { bfr.u[j] = Krow[c0 + j]; bfr.u[4 + j] = Krow[c1 + j]; }
        sacc[t] = __builtin_amdgcn_wmma_f32_16x16x32_bf16(
            false, a.bf, false, bfr.bf, (short)0, sacc[t], false, false);
      }
    }

#pragma unroll
    for (int r = 0; r < 8; ++r) {
      const int qg = q0 + wave * 16 + r + 8 * half;
      float s0 = sacc[0][r] * SCALE;
      float s1 = sacc[1][r] * SCALE;
      if (kv0 + ln      > qg) s0 = -__builtin_inff();
      if (kv0 + 16 + ln > qg) s1 = -__builtin_inff();
      float rm = fmaxf(s0, s1);
#pragma unroll
      for (int sh = 8; sh >= 1; sh >>= 1) rm = fmaxf(rm, __shfl_xor(rm, sh, 32));
      const float mnew = fmaxf(mreg[r], rm);
      float ps = __expf(s0 - mnew) + __expf(s1 - mnew);
#pragma unroll
      for (int sh = 8; sh >= 1; sh >>= 1) ps += __shfl_xor(ps, sh, 32);
      lreg[r] = lreg[r] * __expf(mreg[r] - mnew) + ps;
      mreg[r] = mnew;
    }
    __syncthreads();
  }

  if (ln == 0) {
#pragma unroll
    for (int r = 0; r < 8; ++r) {
      int row = q0 + wave * 16 + r + 8 * half;
      mbuf[(size_t)b * SS + row] = mreg[r];
      lbuf[(size_t)b * SS + row] = lreg[r];
    }
  }
}

// ---------------------------------------------------------------------------
// 4) Pass 2: O = (sum_kv exp(s - m) * V) / l, d-slice of 64. V pre-transposed.
// ---------------------------------------------------------------------------
__global__ __launch_bounds__(128)
void attn_out_kernel(const unsigned short* __restrict__ Q,
                     const unsigned short* __restrict__ Km,
                     const unsigned short* __restrict__ VT,
                     const float* __restrict__ mbuf,
                     const float* __restrict__ lbuf,
                     float* __restrict__ out) {
  __shared__ __align__(16) unsigned int   Kt[32 * (DD / 2)];  // 64 KB
  __shared__ __align__(16) unsigned int   Vp[64 * 16];        // [64 d][16 u32] 4 KB
  __shared__ __align__(16) unsigned short Pt[4][16][32];      // per-wave P, 4 KB

  const int tid  = threadIdx.x;
  const int wave = tid >> 5;
  const int lane = tid & 31;
  const int half = lane >> 4;
  const int ln   = lane & 15;
  const int b    = blockIdx.z;
  const int q0   = blockIdx.y * 64;
  const int d0   = blockIdx.x * 64;

  const int qrow = q0 + wave * 16 + ln;
  const unsigned int* Qrow = (const unsigned int*)Q + ((size_t)b * SS + qrow) * (DD / 2);
  const unsigned int* V32  = (const unsigned int*)VT;

  float mr[8], li[8];
#pragma unroll
  for (int r = 0; r < 8; ++r) {
    int row = q0 + wave * 16 + r + 8 * half;
    mr[r] = mbuf[(size_t)b * SS + row];
    li[r] = 1.0f / lbuf[(size_t)b * SS + row];
  }

  v8f oacc[4] = {};

  const int nkv = ((q0 + 63) >> 5) + 1;
  for (int kt = 0; kt < nkv; ++kt) {
    const int kv0 = kt * 32;
    const unsigned int* Kg = (const unsigned int*)Km + ((size_t)b * SS + kv0) * (DD / 2);
#pragma unroll
    for (int it = 0; it < 32; ++it) {
      int i = (it * 128 + tid) * 4;
      async_b128(lds_off(&Kt[i]), Kg + i);
    }
#pragma unroll
    for (int it = 0; it < 2; ++it) {
      int i = (it * 128 + tid) * 4;
      int row = i >> 4, col = i & 15;
      async_b128(lds_off(&Vp[i]),
                 &V32[((size_t)b * DD + d0 + row) * (SS >> 1) + (kv0 >> 1) + col]);
    }
    wait_async0();
    __syncthreads();

    v8f sacc[2] = {};
    for (int kc = 0; kc < DD; kc += 32) {
      Frag a;
      const int c0 = (kc >> 1) + half * 4;
      const int c1 = (kc >> 1) + 8 + half * 4;
#pragma unroll
      for (int j = 0; j < 4; ++j) { a.u[j] = Qrow[c0 + j]; a.u[4 + j] = Qrow[c1 + j]; }
#pragma unroll
      for (int t = 0; t < 2; ++t) {
        Frag bfr;
        const unsigned int* Krow = &Kt[(t * 16 + ln) * (DD / 2)];
#pragma unroll
        for (int j = 0; j < 4; ++j) { bfr.u[j] = Krow[c0 + j]; bfr.u[4 + j] = Krow[c1 + j]; }
        sacc[t] = __builtin_amdgcn_wmma_f32_16x16x32_bf16(
            false, a.bf, false, bfr.bf, (short)0, sacc[t], false, false);
      }
    }

#pragma unroll
    for (int t = 0; t < 2; ++t)
#pragma unroll
      for (int r = 0; r < 8; ++r) {
        const int qg = q0 + wave * 16 + r + 8 * half;
        float s = sacc[t][r] * SCALE;
        if (kv0 + t * 16 + ln > qg) s = -__builtin_inff();
        Pt[wave][r + 8 * half][t * 16 + ln] = f32_to_bf16(__expf(s - mr[r]));
      }
    __syncthreads();

    Frag pa;
    const unsigned int* Prow = (const unsigned int*)&Pt[wave][ln][0];
#pragma unroll
    for (int j = 0; j < 4; ++j) {
      pa.u[j]     = Prow[half * 4 + j];
      pa.u[4 + j] = Prow[8 + half * 4 + j];
    }
#pragma unroll
    for (int t = 0; t < 4; ++t) {
      Frag vf;
      const unsigned int* Vrow = &Vp[(t * 16 + ln) * 16];
#pragma unroll
      for (int j = 0; j < 4; ++j) {
        vf.u[j]     = Vrow[half * 4 + j];
        vf.u[4 + j] = Vrow[8 + half * 4 + j];
      }
      oacc[t] = __builtin_amdgcn_wmma_f32_16x16x32_bf16(
          false, pa.bf, false, vf.bf, (short)0, oacc[t], false, false);
    }
    __syncthreads();
  }

#pragma unroll
  for (int t = 0; t < 4; ++t)
#pragma unroll
    for (int r = 0; r < 8; ++r) {
      int row = q0 + wave * 16 + r + 8 * half;
      int col = d0 + t * 16 + ln;
      out[((size_t)b * SS + row) * DD + col] = oacc[t][r] * li[r];
    }
}

// ---------------------------------------------------------------------------
extern "C" void kernel_launch(void* const* d_in, const int* in_sizes, int n_in,
                              void* d_out, int out_size, void* d_ws, size_t ws_size,
                              hipStream_t stream) {
  const float* x  = (const float*)d_in[0];
  const float* Wq = (const float*)d_in[1];
  const float* Wk = (const float*)d_in[2];
  const float* Wv = (const float*)d_in[3];
  float* out = (float*)d_out;

  const size_t nx = (size_t)BB * SS * DD;
  const size_t nw = (size_t)DD * DD;

  char* ws = (char*)d_ws;
  unsigned short* xb  = (unsigned short*)ws;  ws += nx * 2;
  unsigned short* wqt = (unsigned short*)ws;  ws += nw * 2;
  unsigned short* wkt = (unsigned short*)ws;  ws += nw * 2;
  unsigned short* wvt = (unsigned short*)ws;  ws += nw * 2;
  unsigned short* qb  = (unsigned short*)ws;  ws += nx * 2;
  unsigned short* kb  = (unsigned short*)ws;  ws += nx * 2;
  unsigned short* vb  = (unsigned short*)ws;  ws += nx * 2;
  unsigned short* vtb = (unsigned short*)ws;  ws += nx * 2;
  float* mbuf = (float*)ws;                   ws += (size_t)BB * SS * 4;
  float* lbuf = (float*)ws;

  cvt_f32_bf16_kernel<<<(int)((nx + 255) / 256), 256, 0, stream>>>(x, xb, (int)nx);
  dim3 tb(32, 8), tgw(DD / 32, DD / 32);
  cvt_transpose_bf16_kernel<<<tgw, tb, 0, stream>>>(Wq, wqt, DD, DD);
  cvt_transpose_bf16_kernel<<<tgw, tb, 0, stream>>>(Wk, wkt, DD, DD);
  cvt_transpose_bf16_kernel<<<tgw, tb, 0, stream>>>(Wv, wvt, DD, DD);

  dim3 gg(DD / 64, (BB * SS) / 128);
  gemm_bf16_kernel<<<gg, 256, 0, stream>>>(xb, wqt, qb, BB * SS, DD, DD);
  gemm_bf16_kernel<<<gg, 256, 0, stream>>>(xb, wkt, kb, BB * SS, DD, DD);
  gemm_bf16_kernel<<<gg, 256, 0, stream>>>(xb, wvt, vb, BB * SS, DD, DD);

  dim3 tgv(DD / 32, SS / 32, BB);
  transpose_bf16_kernel<<<tgv, tb, 0, stream>>>(vb, vtb, SS, DD);

  dim3 gs(SS / 128, BB);
  attn_stats_kernel<<<gs, 256, 0, stream>>>(qb, kb, mbuf, lbuf);

  dim3 go(DD / 64, SS / 64, BB);
  attn_out_kernel<<<go, 128, 0, stream>>>(qb, kb, vtb, mbuf, lbuf, out);
}
